// AdditiveAttention_26912265077414
// MI455X (gfx1250) — matrix operand order repaired and checked by reading, not compiled, not run on hardware
//
#include <hip/hip_runtime.h>
#include <hip/hip_bf16.h>
#include <math.h>

// Problem constants (match reference)
#define NB   32      // batch
#define TK   8192    // timesteps
#define DD   256     // feature dim (keys/values/queries)
#define HH   256     // hidden dim
#define CH   32      // chunks of Tk per batch
#define ROWS_PER_BLOCK (TK / CH)            // 256
#define TILES_PER_BLOCK (ROWS_PER_BLOCK/16) // 16
#define NWAVES 8
#define TILES_PER_WAVE (TILES_PER_BLOCK / NWAVES) // 2

typedef __attribute__((ext_vector_type(16))) __bf16       v16bf;
typedef __attribute__((ext_vector_type(8)))  float        v8f;
typedef __attribute__((ext_vector_type(4)))  float        f32x4;
typedef __attribute__((ext_vector_type(4)))  unsigned int u32x4;

union FragB { u32x4 u[2]; v16bf v; };

#define LOG2E  1.4426950408889634f
#define LOG2E2 2.8853900817779268f   // 2*log2(e)

// Native-trans tanh: one v_tanh_f32 if the builtin exists, else branch-free
// v_exp_f32 + v_rcp_f32 formulation (saturates correctly at +/-1 via inf/0).
static __device__ __forceinline__ float fast_tanh(float x) {
#if __has_builtin(__builtin_amdgcn_tanhf)
    return __builtin_amdgcn_tanhf(x);
#elif __has_builtin(__builtin_amdgcn_tanh_f32)
    return __builtin_amdgcn_tanh_f32(x);
#else
    const float e = __builtin_amdgcn_exp2f(x * LOG2E2);  // e^(2x)
    return 1.f - 2.f * __builtin_amdgcn_rcpf(e + 1.f);
#endif
}

// exp(x) for x <= 0 (softmax-safe): single v_exp_f32, no range reduction.
static __device__ __forceinline__ float fast_exp(float x) {
#if __has_builtin(__builtin_amdgcn_exp2f)
    return __builtin_amdgcn_exp2f(x * LOG2E);
#else
    return __expf(x);
#endif
}

// ---------------------------------------------------------------------------
// Kernel 1: q = queries @ W_q   (32x256) — tiny, scalar
// ---------------------------------------------------------------------------
__global__ void qproj_kernel(const float* __restrict__ queries,
                             const float* __restrict__ W_q,
                             float* __restrict__ q_out) {
    const int b = blockIdx.x;
    const int h = threadIdx.x;
    const float* qrow = queries + b * DD;
    float acc = 0.f;
    for (int d = 0; d < DD; ++d)
        acc = __builtin_fmaf(qrow[d], W_q[d * HH + h], acc);
    q_out[b * HH + h] = acc;
}

// ---------------------------------------------------------------------------
// Kernel 2: per-(batch, Tk-chunk) partial:
//   k = keys_chunk @ W_k  (bf16 WMMA, f32 accum)
//   scores = tanh(q + k) . w_v ; online softmax ; o_partial = sum w * values
// ---------------------------------------------------------------------------
__global__ __launch_bounds__(256)
void addattn_partial_kernel(const float* __restrict__ keys,
                            const float* __restrict__ values,
                            const float* __restrict__ W_k,
                            const float* __restrict__ w_v,
                            const float* __restrict__ q,     // (NB,HH)
                            float* __restrict__ m_part,      // (NB*CH)
                            float* __restrict__ s_part,      // (NB*CH)
                            float* __restrict__ o_part) {    // (NB*CH*DD)
    extern __shared__ __align__(16) char smem_raw[];
    __bf16* wk_lds = (__bf16*)smem_raw;                       // 16*8*32*16 bf16 = 128 KB
    float*  q_lds  = (float*)(smem_raw + (HH/16)*8*32*16*2);  // 256 f
    float*  wv_lds = q_lds + HH;                              // 256 f
    float*  o_acc  = wv_lds + HH;                             // 256 f
    float*  sc_lds = o_acc + DD;                              // 8*16 f
    float*  wm     = sc_lds + NWAVES * 16;                    // 8 f
    float*  wsum   = wm + NWAVES;                             // 8 f

    const int chunk = blockIdx.x;
    const int b     = blockIdx.y;
    const int tid   = threadIdx.x;
    const int lane  = tid & 31;
    const int wave  = tid >> 5;

    // ---- Stage W_k (f32, row-major d x h) into LDS as bf16 B-fragments.
    // B-fragment (K=32 x N=16) layout: lane = nl + 16*(k32>=16), elem j = k32 & 15.
    for (int e = tid; e < DD * HH; e += 256) {
        const int d   = e >> 8;     // K-dim index 0..255 (coalesced h inner)
        const int h   = e & 255;    // N-dim index
        const float val = W_k[e];
        const int kk  = d >> 5;     // K step (0..7)
        const int k32 = d & 31;
        const int nt  = h >> 4;     // N tile (0..15)
        const int nl  = h & 15;
        const int fl  = nl + (k32 & 16);
        const int j   = k32 & 15;
        wk_lds[(((nt * 8 + kk) * 32 + fl) << 4) + j] = (__bf16)val;
    }
    if (tid < HH) {
        q_lds[tid]  = q[b * HH + tid];
        wv_lds[tid] = w_v[tid];
        o_acc[tid]  = 0.f;
    }
    __syncthreads();

    float m_run = -INFINITY, s_run = 0.f;
    float o_run[8];
    #pragma unroll
    for (int c = 0; c < 8; ++c) o_run[c] = 0.f;

    const int t_base = chunk * ROWS_PER_BLOCK;
    const int hl = lane & 15;

    for (int it = 0; it < TILES_PER_WAVE; ++it) {
        const int t0 = t_base + (wave + NWAVES * it) * 16;

        // ---- A fragments: 16 key rows x 256 d -> 8 frags of 16x32 bf16.
        // lane<16: row=lane, K = {d0..d0+15}, d0 = kk*32
        // lane>=16: same rows, d0 = kk*32 + 16
        v16bf a[8];
        {
            const int row = t0 + hl;
            const size_t roff = ((size_t)b * TK + row) * DD;
            const int dhalf = (lane >> 4) << 4;  // 0 or 16
            #pragma unroll
            for (int kk = 0; kk < 8; ++kk) {
                const int d0 = kk * 32 + dhalf;
                const f32x4* p  = (const f32x4*)(keys + roff + d0);
                const f32x4* p2 = (const f32x4*)(keys + roff + d0 + 8);
                f32x4 f0 = p[0],  f1 = p[1];
                f32x4 f2 = p2[0], f3 = p2[1];
                #pragma unroll
                for (int u = 0; u < 4; ++u) {
                    a[kk][u]      = (__bf16)f0[u];
                    a[kk][4 + u]  = (__bf16)f1[u];
                    a[kk][8 + u]  = (__bf16)f2[u];
                    a[kk][12 + u] = (__bf16)f3[u];
                }
            }
        }

        // ---- K-projection + fused tanh(q+k).w_v partial scores
        // Two N-tiles in flight -> two independent WMMA accumulator chains.
        float sp[8];
        #pragma unroll
        for (int i = 0; i < 8; ++i) sp[i] = 0.f;

        for (int nt = 0; nt < HH / 16; nt += 2) {
            v8f c0 = {}, c1 = {};
            #pragma unroll
            for (int kk = 0; kk < 8; ++kk) {
                FragB fb0, fb1;
                const u32x4* bp0 =
                    (const u32x4*)&wk_lds[(((nt + 0) * 8 + kk) * 32 + lane) << 4];
                const u32x4* bp1 =
                    (const u32x4*)&wk_lds[(((nt + 1) * 8 + kk) * 32 + lane) << 4];
                fb0.u[0] = bp0[0]; fb0.u[1] = bp0[1];
                fb1.u[0] = bp1[0]; fb1.u[1] = bp1[1];
                c0 = __builtin_amdgcn_wmma_f32_16x16x32_bf16(
                        false, a[kk], false, fb0.v, (short)0, c0, false, false);
                c1 = __builtin_amdgcn_wmma_f32_16x16x32_bf16(
                        false, a[kk], false, fb1.v, (short)0, c1, false, false);
            }
            // C layout: VGPR i, lanes 0-15: (M=i, N=lane); lanes 16-31: (M=8+i, N=lane-16)
            const int h0 = (nt + 0) * 16 + hl;
            const int h1 = (nt + 1) * 16 + hl;
            const float qv0 = q_lds[h0],  qv1 = q_lds[h1];
            const float wv0 = wv_lds[h0], wv1 = wv_lds[h1];
            #pragma unroll
            for (int i = 0; i < 8; ++i) {
                sp[i] = __builtin_fmaf(fast_tanh(c0[i] + qv0), wv0, sp[i]);
                sp[i] = __builtin_fmaf(fast_tanh(c1[i] + qv1), wv1, sp[i]);
            }
        }

        // ---- reduce partial scores across each 16-lane half
        #pragma unroll
        for (int off = 1; off < 16; off <<= 1) {
            #pragma unroll
            for (int i = 0; i < 8; ++i)
                sp[i] += __shfl_xor(sp[i], off, 32);
        }
        // lanes 0-15 hold rows 0..7, lanes 16-31 hold rows 8..15; publish
        if (lane == 0) {
            #pragma unroll
            for (int i = 0; i < 8; ++i) sc_lds[wave * 16 + i] = sp[i];
        }
        if (lane == 16) {
            #pragma unroll
            for (int i = 0; i < 8; ++i) sc_lds[wave * 16 + 8 + i] = sp[i];
        }
        __builtin_amdgcn_wave_barrier();

        float sc[16];
        #pragma unroll
        for (int r = 0; r < 16; ++r) sc[r] = sc_lds[wave * 16 + r];

        // ---- online softmax update for this 16-row tile
        float m_t = sc[0];
        #pragma unroll
        for (int r = 1; r < 16; ++r) m_t = fmaxf(m_t, sc[r]);
        const float m_new = fmaxf(m_run, m_t);
        const float scale = fast_exp(m_run - m_new);
        float w[16], s_add = 0.f;
        #pragma unroll
        for (int r = 0; r < 16; ++r) { w[r] = fast_exp(sc[r] - m_new); s_add += w[r]; }
        s_run = s_run * scale + s_add;
        #pragma unroll
        for (int c2 = 0; c2 < 8; ++c2) o_run[c2] *= scale;

        const int col0 = lane * 8;
        #pragma unroll
        for (int r = 0; r < 16; ++r) {
            const size_t voff = ((size_t)b * TK + t0 + r) * DD + col0;
            const f32x4* vp = (const f32x4*)(values + voff);
            f32x4 v0 = vp[0], v1 = vp[1];
            #pragma unroll
            for (int u = 0; u < 4; ++u) {
                o_run[u]     = __builtin_fmaf(w[r], v0[u], o_run[u]);
                o_run[4 + u] = __builtin_fmaf(w[r], v1[u], o_run[4 + u]);
            }
        }
        m_run = m_new;
    }

    // ---- combine the 8 wave partials into one block partial
    if (lane == 0) { wm[wave] = m_run; wsum[wave] = s_run; }
    __syncthreads();
    float m_blk = wm[0];
    #pragma unroll
    for (int v2 = 1; v2 < NWAVES; ++v2) m_blk = fmaxf(m_blk, wm[v2]);
    const float factor = fast_exp(m_run - m_blk);
    #pragma unroll
    for (int c2 = 0; c2 < 8; ++c2)
        atomicAdd(&o_acc[lane * 8 + c2], o_run[c2] * factor);  // ds_add_f32
    float s_blk = 0.f;
    #pragma unroll
    for (int v2 = 0; v2 < NWAVES; ++v2)
        s_blk += wsum[v2] * fast_exp(wm[v2] - m_blk);
    __syncthreads();

    const int pidx = b * CH + chunk;
    o_part[(size_t)pidx * DD + tid] = o_acc[tid];
    if (tid == 0) { m_part[pidx] = m_blk; s_part[pidx] = s_blk; }
}

// ---------------------------------------------------------------------------
// Kernel 3: reduce CH partials per batch -> output (NB x DD)
// ---------------------------------------------------------------------------
__global__ void addattn_finalize_kernel(const float* __restrict__ m_part,
                                        const float* __restrict__ s_part,
                                        const float* __restrict__ o_part,
                                        float* __restrict__ out) {
    const int b = blockIdx.x;
    const int h = threadIdx.x;
    float m_fin = -INFINITY;
    for (int i = 0; i < CH; ++i) m_fin = fmaxf(m_fin, m_part[b * CH + i]);
    float s_fin = 0.f, acc = 0.f;
    for (int i = 0; i < CH; ++i) {
        const float f = fast_exp(m_part[b * CH + i] - m_fin);
        s_fin += s_part[b * CH + i] * f;
        acc   += o_part[((size_t)(b * CH + i)) * DD + h] * f;
    }
    out[b * DD + h] = acc / s_fin;
}

// ---------------------------------------------------------------------------
extern "C" void kernel_launch(void* const* d_in, const int* in_sizes, int n_in,
                              void* d_out, int out_size, void* d_ws, size_t ws_size,
                              hipStream_t stream) {
    const float* queries = (const float*)d_in[0];  // (NB, DD)
    const float* keys    = (const float*)d_in[1];  // (NB, TK, DD)
    const float* values  = (const float*)d_in[2];  // (NB, TK, DD)
    const float* W_q     = (const float*)d_in[3];  // (DD, HH)
    const float* W_k     = (const float*)d_in[4];  // (DD, HH)
    const float* w_v     = (const float*)d_in[5];  // (HH)
    float* out = (float*)d_out;

    // workspace carve-up (floats)
    float* ws = (float*)d_ws;
    float* q_ws    = ws;                       // NB*HH
    float* m_part  = q_ws + NB * HH;           // NB*CH
    float* s_part  = m_part + NB * CH;         // NB*CH
    float* o_part  = s_part + NB * CH;         // NB*CH*DD

    const size_t lds_bytes =
        (size_t)(HH / 16) * 8 * 32 * 16 * 2           // W_k bf16 fragments
        + (HH + HH + DD + NWAVES * 16 + 2 * NWAVES) * sizeof(float);

    static bool attr_set = false;  // idempotent host-side attribute, not a stream op
    if (!attr_set) {
        hipFuncSetAttribute((const void*)addattn_partial_kernel,
                            hipFuncAttributeMaxDynamicSharedMemorySize,
                            (int)lds_bytes);
        attr_set = true;
    }

    qproj_kernel<<<dim3(NB), dim3(HH), 0, stream>>>(queries, W_q, q_ws);
    addattn_partial_kernel<<<dim3(CH, NB), dim3(256), lds_bytes, stream>>>(
        keys, values, W_k, w_v, q_ws, m_part, s_part, o_part);
    addattn_finalize_kernel<<<dim3(NB), dim3(DD), 0, stream>>>(
        m_part, s_part, o_part, out);
}
